// EncoderBlock_4432406249660
// MI455X (gfx1250) — matrix operand order repaired and checked by reading, compiled, not run on hardware
//
#include <hip/hip_runtime.h>

// ---------------------------------------------------------------------------
// Types for CDNA5 WMMA (wave32): A/B = v16h (16 halves = 8 VGPRs), C/D = v8f.
// ---------------------------------------------------------------------------
typedef __attribute__((ext_vector_type(16))) _Float16 v16h;
typedef __attribute__((ext_vector_type(8)))  float    v8f;
typedef __attribute__((ext_vector_type(4)))  int      v4i;

struct F16x16 { union { v16h v; uint4 q[2]; _Float16 h[16]; }; };
struct H8     { union { uint4 q; _Float16 h[8]; }; };

__device__ __forceinline__ v8f zero_v8f() {
    v8f z;
#pragma unroll
    for (int i = 0; i < 8; ++i) z[i] = 0.0f;
    return z;
}

#define DM   1024    // d_model
#define DFF  4096
#define SEQ  2048
#define NH   16
#define DK   64

// gfx1250 async global->LDS path (ASYNCcnt), if the toolchain exposes it.
// Builtin signature (from hipcc diagnostic): (v4i AS1*, v4i AS3*, imm, cpol).
#if __has_builtin(__builtin_amdgcn_global_load_async_to_lds_b128) && \
    __has_builtin(__builtin_amdgcn_s_wait_asynccnt)
#define USE_ASYNC_LDS 1
typedef __attribute__((address_space(1))) v4i* gptr_v4i;
typedef __attribute__((address_space(3))) v4i* lptr_v4i;
__device__ __forceinline__ void cp16(const _Float16* g, _Float16* l) {
    __builtin_amdgcn_global_load_async_to_lds_b128((gptr_v4i)g, (lptr_v4i)l, 0, 0);
}
#else
__device__ __forceinline__ void cp16(const _Float16* g, _Float16* l) {
    *(uint4*)l = *(const uint4*)g;
}
#endif

// ---------------------------------------------------------------------------
// fp32 -> fp16 conversion (weights)
// ---------------------------------------------------------------------------
__global__ void cvt_f32_f16(const float* __restrict__ src, _Float16* __restrict__ dst, int n) {
    int i = blockIdx.x * blockDim.x + threadIdx.x;
    for (; i < n; i += gridDim.x * blockDim.x) dst[i] = (_Float16)src[i];
}

// ---------------------------------------------------------------------------
// LayerNorm, torch variant: unbiased std (ddof=1), eps added to std.
// ---------------------------------------------------------------------------
__global__ void ln_kernel(const float* __restrict__ x,
                          const float* __restrict__ gamma,
                          const float* __restrict__ beta,
                          _Float16* __restrict__ out) {
    const int row = blockIdx.x, t = threadIdx.x;
    __shared__ float s1[256], s2[256];
    const float* xr = x + (size_t)row * DM;
    float sum = 0.f, sq = 0.f;
    for (int i = t; i < DM; i += 256) { float v = xr[i]; sum += v; sq += v * v; }
    s1[t] = sum; s2[t] = sq; __syncthreads();
    for (int o = 128; o > 0; o >>= 1) {
        if (t < o) { s1[t] += s1[t + o]; s2[t] += s2[t + o]; }
        __syncthreads();
    }
    const float mean = s1[0] * (1.0f / DM);
    const float var  = fmaxf((s2[0] - (float)DM * mean * mean) * (1.0f / (DM - 1)), 0.f);
    const float inv  = 1.0f / (sqrtf(var) + 1e-6f);
    for (int i = t; i < DM; i += 256) {
        float v = (xr[i] - mean) * inv * gamma[i] + beta[i];
        out[(size_t)row * DM + i] = (_Float16)v;
    }
}

// ---------------------------------------------------------------------------
// Tiled WMMA GEMM: C[M,N] = A[M,K] @ B[N,K]^T (+bias)(+res)(+relu)
// 128 threads = 4 waves; block tile 128x128; each wave owns a 64x64 quadrant
// (4 A-frags x 4 B-frags = 16 WMMA per k-step, 1 ds_load_b128 per WMMA).
// Double-buffered global->LDS copies (async DMA when available).
// ---------------------------------------------------------------------------
template <bool BIAS, bool RELU, bool RES, typename OutT>
__launch_bounds__(128)
__global__ void gemm_wmma(const _Float16* __restrict__ A,
                          const _Float16* __restrict__ B,
                          OutT* __restrict__ C,
                          const float* __restrict__ bias,
                          const float* __restrict__ res,
                          int M, int N, int K) {
    __shared__ _Float16 Ash[2][128 * 40];
    __shared__ _Float16 Bsh[2][128 * 40];
    const int t = threadIdx.x;
    const int wave = t >> 5, lane = t & 31, g = lane >> 4, n15 = lane & 15;
    const int m0 = blockIdx.y * 128, n0 = blockIdx.x * 128;
    const int wm = (wave & 1) * 64, wn = (wave >> 1) * 64;

    v8f acc[4][4];
#pragma unroll
    for (int i = 0; i < 4; ++i)
#pragma unroll
        for (int j = 0; j < 4; ++j) acc[i][j] = zero_v8f();

    auto copy_tile = [&](int buf, int k0) {
#pragma unroll
        for (int p = 0; p < 4; ++p) {
            const int c = t + 128 * p, rr = c >> 2, seg = c & 3;
            cp16(&A[(size_t)(m0 + rr) * K + k0 + seg * 8], &Ash[buf][rr * 40 + seg * 8]);
        }
#pragma unroll
        for (int p = 0; p < 4; ++p) {
            const int c = t + 128 * p, rr = c >> 2, seg = c & 3;
            cp16(&B[(size_t)(n0 + rr) * K + k0 + seg * 8], &Bsh[buf][rr * 40 + seg * 8]);
        }
    };

    copy_tile(0, 0);
    int cur = 0;
    for (int k0 = 0; k0 < K; k0 += 32) {
#ifdef USE_ASYNC_LDS
        __builtin_amdgcn_s_wait_asynccnt(0);  // this tile's DMA landed in LDS
#endif
        __syncthreads();                      // ...for every wave in the block
        if (k0 + 32 < K) copy_tile(cur ^ 1, k0 + 32);

        F16x16 af[4];
#pragma unroll
        for (int i = 0; i < 4; ++i) {
            const int r = wm + i * 16 + n15;
            af[i].q[0] = *(const uint4*)&Ash[cur][r * 40 + g * 8];
            af[i].q[1] = *(const uint4*)&Ash[cur][r * 40 + 16 + g * 8];
        }
#pragma unroll
        for (int j = 0; j < 4; ++j) {
            F16x16 bf;
            const int r = wn + j * 16 + n15;
            bf.q[0] = *(const uint4*)&Bsh[cur][r * 40 + g * 8];
            bf.q[1] = *(const uint4*)&Bsh[cur][r * 40 + 16 + g * 8];
#pragma unroll
            for (int i = 0; i < 4; ++i)
                acc[i][j] = __builtin_amdgcn_wmma_f32_16x16x32_f16(
                    false, af[i].v, false, bf.v, (short)0, acc[i][j], false, false);
        }
        __syncthreads();                      // reads done before buffer reuse
        cur ^= 1;
    }

#pragma unroll
    for (int j = 0; j < 4; ++j) {
        const int col = n0 + wn + j * 16 + n15;
        const float bv = BIAS ? bias[col] : 0.f;
#pragma unroll
        for (int i = 0; i < 4; ++i) {
#pragma unroll
            for (int r = 0; r < 8; ++r) {
                const int row = m0 + wm + i * 16 + r + g * 8;
                float v = acc[i][j][r] + bv;
                if (RES) v += res[(size_t)row * N + col];
                if (RELU) v = fmaxf(v, 0.f);
                C[(size_t)row * N + col] = (OutT)v;
            }
        }
    }
}

// ---------------------------------------------------------------------------
// Flash attention, transposed-score formulation.
//   S^T[32x16] = K[32x64] @ Q^T          -> each QUERY is a lane column:
//   all 32 key-scores of a query live in-lane (+- the lane^16 half), so
//   softmax max/sum are 15 VALU ops + ONE shfl_xor(16) each; m/l/alpha are
//   per-lane scalars, and O is kept transposed: O^T += V^T @ P^T.
// Block = 4 waves sharing one (b,h); K/V tiles staged once per block.
// ---------------------------------------------------------------------------
__launch_bounds__(128)
__global__ void flash_kernel(const _Float16* __restrict__ Q,
                             const _Float16* __restrict__ Km,
                             const _Float16* __restrict__ Vm,
                             const int* __restrict__ mask,
                             _Float16* __restrict__ Out) {
    // halves: Ks 32x72 (2304) | Vt 64x40 (2560) | Qs 4x16x72 (4608)
    //         | Pt 4x16x40 (2560) | Mb 32
    __shared__ _Float16 sm[12096];
    const int t = threadIdx.x;
    const int wave = t >> 5, lane = t & 31, g = lane >> 4, n15 = lane & 15;
    _Float16* Ks = sm;
    _Float16* Vt = sm + 2304;
    _Float16* Qs = sm + 4864 + wave * 1152;
    _Float16* Pt = sm + 9472 + wave * 640;
    _Float16* Mb = sm + 12032;

    const int bh = blockIdx.y, b = bh >> 4, h = bh & 15;
    const int q0 = blockIdx.x * 64 + wave * 16;
    const int hofs = h * DK;
    const float scale = 0.125f;  // 1/sqrt(64)

    // stage Q tile (per wave, 16 rows x 64 halves)
    {
        const int qr = lane >> 1, half = lane & 1;
        const size_t gq = (size_t)(b * SEQ + q0 + qr) * DM + hofs;
#pragma unroll
        for (int c = 0; c < 4; ++c) {
            const int seg = half * 4 + c;
            *(uint4*)&Qs[qr * 72 + seg * 8] = *(const uint4*)&Q[gq + seg * 8];
        }
    }

    float m_q = -1e30f, l_q = 0.f;   // per-lane scalars (lane = query)
    v8f acc[4];                      // O^T: rows d=i*16+r+8g, col = query n15
#pragma unroll
    for (int i = 0; i < 4; ++i) acc[i] = zero_v8f();

    for (int kb = 0; kb < SEQ; kb += 32) {
        __syncthreads();  // previous iteration's reads done (also fences Qs once)

        // ---- stage K tile 32x64 (block-wide, 2x16B chunks per thread)
#pragma unroll
        for (int p = 0; p < 2; ++p) {
            const int c = t + 128 * p, rr = c >> 3, seg = c & 7;
            *(uint4*)&Ks[rr * 72 + seg * 8] =
                *(const uint4*)&Km[(size_t)(b * SEQ + kb + rr) * DM + hofs + seg * 8];
        }
        // ---- stage V tile transposed: Vt[d][key] (block-wide)
        {
            const int kr = t >> 2, qt = t & 3;
            const size_t vrow = (size_t)(b * SEQ + kb + kr) * DM + hofs;
#pragma unroll
            for (int c2 = 0; c2 < 2; ++c2) {
                H8 u; u.q = *(const uint4*)&Vm[vrow + qt * 16 + c2 * 8];
#pragma unroll
                for (int i = 0; i < 8; ++i)
                    Vt[(qt * 16 + c2 * 8 + i) * 40 + kr] = u.h[i];
            }
        }
        // ---- per-key additive mask bias (f16): 0 or -65504
        if (t < 32)
            Mb[t] = mask[b * SEQ + kb + t] ? (_Float16)0.f : (_Float16)(-65504.f);
        __syncthreads();

        // ---- S^T[32x16] = K[32x64] @ Q^T : rows=keys, cols=queries(lanes)
        v8f cs[2];
        cs[0] = zero_v8f(); cs[1] = zero_v8f();
        {
            F16x16 Qb0, Qb1;   // B operand: Q^T (d x query)
            Qb0.q[0] = *(const uint4*)&Qs[n15 * 72 + g * 8];
            Qb0.q[1] = *(const uint4*)&Qs[n15 * 72 + 16 + g * 8];
            Qb1.q[0] = *(const uint4*)&Qs[n15 * 72 + 32 + g * 8];
            Qb1.q[1] = *(const uint4*)&Qs[n15 * 72 + 32 + 16 + g * 8];
#pragma unroll
            for (int j = 0; j < 2; ++j) {
                const int base = (j * 16 + n15) * 72;
                F16x16 ka0, ka1;   // A operand: K rows
                ka0.q[0] = *(const uint4*)&Ks[base + g * 8];
                ka0.q[1] = *(const uint4*)&Ks[base + 16 + g * 8];
                ka1.q[0] = *(const uint4*)&Ks[base + 32 + g * 8];
                ka1.q[1] = *(const uint4*)&Ks[base + 32 + 16 + g * 8];
                cs[j] = __builtin_amdgcn_wmma_f32_16x16x32_f16(false, ka0.v, false, Qb0.v,
                                                               (short)0, cs[j], false, false);
                cs[j] = __builtin_amdgcn_wmma_f32_16x16x32_f16(false, ka1.v, false, Qb1.v,
                                                               (short)0, cs[j], false, false);
            }
        }

        // ---- in-lane softmax over this lane's 16 keys (+ lane^16 half)
        H8 mb0, mb1;   // bias for keys [8g,8g+8) and [16+8g,16+8g+8)
        mb0.q = *(const uint4*)&Mb[8 * g];
        mb1.q = *(const uint4*)&Mb[16 + 8 * g];
        float s_v[2][8];
        float tmax = -1e30f;
#pragma unroll
        for (int r = 0; r < 8; ++r) {
            s_v[0][r] = fmaf(cs[0][r], scale, (float)mb0.h[r]);
            s_v[1][r] = fmaf(cs[1][r], scale, (float)mb1.h[r]);
            tmax = fmaxf(tmax, fmaxf(s_v[0][r], s_v[1][r]));
        }
        tmax = fmaxf(tmax, __shfl_xor(tmax, 16, 32));
        const float mn = fmaxf(m_q, tmax);
        const float al = __expf(m_q - mn);   // underflows to 0 at sentinel
        m_q = mn;
        float ps = 0.f;
        H8 p0, p1;
#pragma unroll
        for (int r = 0; r < 8; ++r) {
            const float e0 = __expf(s_v[0][r] - mn);
            const float e1 = __expf(s_v[1][r] - mn);
            ps += e0 + e1;
            p0.h[r] = (_Float16)e0;
            p1.h[r] = (_Float16)e1;
        }
        ps += __shfl_xor(ps, 16, 32);
        l_q = l_q * al + ps;
#pragma unroll
        for (int i = 0; i < 4; ++i)
#pragma unroll
            for (int r = 0; r < 8; ++r) acc[i][r] *= al;

        // ---- store P^T row (query n15): 2 packed b128 stores
        *(uint4*)&Pt[n15 * 40 + 8 * g]      = p0.q;
        *(uint4*)&Pt[n15 * 40 + 16 + 8 * g] = p1.q;
        __syncthreads();   // P^T visible across the wave halves

        // ---- O^T[64x16] += V^T[64x32] @ P^T[32x16]
        F16x16 pb;
        pb.q[0] = *(const uint4*)&Pt[n15 * 40 + g * 8];
        pb.q[1] = *(const uint4*)&Pt[n15 * 40 + 16 + g * 8];
#pragma unroll
        for (int i = 0; i < 4; ++i) {
            F16x16 va;
            const int base = (i * 16 + n15) * 40;
            va.q[0] = *(const uint4*)&Vt[base + g * 8];
            va.q[1] = *(const uint4*)&Vt[base + 16 + g * 8];
            acc[i] = __builtin_amdgcn_wmma_f32_16x16x32_f16(
                false, va.v, false, pb.v, (short)0, acc[i], false, false);
        }
    }

    // ---- epilogue: divide by l, packed b128 stores (lane = query row)
    const float inv = (l_q > 0.f) ? 1.0f / l_q : 0.f;
    const size_t orow = (size_t)(b * SEQ + q0 + n15) * DM + hofs;
#pragma unroll
    for (int i = 0; i < 4; ++i) {
        H8 o;
#pragma unroll
        for (int r = 0; r < 8; ++r) o.h[r] = (_Float16)(acc[i][r] * inv);
        *(uint4*)&Out[orow + i * 16 + 8 * g] = o.q;
    }
}

// ---------------------------------------------------------------------------
// Host-side orchestration
// ---------------------------------------------------------------------------
extern "C" void kernel_launch(void* const* d_in, const int* in_sizes, int n_in,
                              void* d_out, int out_size, void* d_ws, size_t ws_size,
                              hipStream_t stream) {
    const float* x     = (const float*)d_in[0];
    const int*   smask = (const int*)  d_in[1];
    const float* w_q   = (const float*)d_in[2];
    const float* w_k   = (const float*)d_in[3];
    const float* w_v   = (const float*)d_in[4];
    const float* w_o   = (const float*)d_in[5];
    const float* w1    = (const float*)d_in[6];
    const float* b1    = (const float*)d_in[7];
    const float* w2    = (const float*)d_in[8];
    const float* b2    = (const float*)d_in[9];
    const float* gamma1 = (const float*)d_in[10];
    const float* beta1  = (const float*)d_in[11];
    const float* gamma2 = (const float*)d_in[12];
    const float* beta2  = (const float*)d_in[13];
    float* out = (float*)d_out;

    const int M = in_sizes[0] / DM;          // B*S = 8192
    char* ws = (char*)d_ws;
    const size_t MB = 1024 * 1024;
    _Float16* wq16 = (_Float16*)(ws + 0 * MB);
    _Float16* wk16 = (_Float16*)(ws + 2 * MB);
    _Float16* wv16 = (_Float16*)(ws + 4 * MB);
    _Float16* wo16 = (_Float16*)(ws + 6 * MB);
    _Float16* w116 = (_Float16*)(ws + 8 * MB);
    _Float16* w216 = (_Float16*)(ws + 16 * MB);
    _Float16* hln  = (_Float16*)(ws + 24 * MB);   // reused for LN1 then LN2
    _Float16* qb   = (_Float16*)(ws + 40 * MB);
    _Float16* kb   = (_Float16*)(ws + 56 * MB);
    _Float16* vb   = (_Float16*)(ws + 72 * MB);
    _Float16* attn = (_Float16*)(ws + 88 * MB);
    float*    x2   = (float*)   (ws + 104 * MB);  // 32 MB
    _Float16* ffn1 = qb;                           // aliases dead q/k/v/attn (64 MB)

    cvt_f32_f16<<<4096, 256, 0, stream>>>(w_q, wq16, DM * DM);
    cvt_f32_f16<<<4096, 256, 0, stream>>>(w_k, wk16, DM * DM);
    cvt_f32_f16<<<4096, 256, 0, stream>>>(w_v, wv16, DM * DM);
    cvt_f32_f16<<<4096, 256, 0, stream>>>(w_o, wo16, DM * DM);
    cvt_f32_f16<<<8192, 256, 0, stream>>>(w1, w116, DFF * DM);
    cvt_f32_f16<<<8192, 256, 0, stream>>>(w2, w216, DM * DFF);

    ln_kernel<<<M, 256, 0, stream>>>(x, gamma1, beta1, hln);

    dim3 gqkv(DM / 128, M / 128);
    gemm_wmma<false, false, false, _Float16><<<gqkv, 128, 0, stream>>>(
        hln, wq16, qb, nullptr, nullptr, M, DM, DM);
    gemm_wmma<false, false, false, _Float16><<<gqkv, 128, 0, stream>>>(
        hln, wk16, kb, nullptr, nullptr, M, DM, DM);
    gemm_wmma<false, false, false, _Float16><<<gqkv, 128, 0, stream>>>(
        hln, wv16, vb, nullptr, nullptr, M, DM, DM);

    dim3 gfa(SEQ / 64, (M / SEQ) * NH);
    flash_kernel<<<gfa, 128, 0, stream>>>(qb, kb, vb, smask, attn);

    gemm_wmma<false, false, true, float><<<gqkv, 128, 0, stream>>>(
        attn, wo16, x2, nullptr, x, M, DM, DM);

    ln_kernel<<<M, 256, 0, stream>>>(x2, gamma2, beta2, hln);

    dim3 gff1(DFF / 128, M / 128);
    gemm_wmma<true, true, false, _Float16><<<gff1, 128, 0, stream>>>(
        hln, w116, ffn1, b1, nullptr, M, DFF, DM);

    gemm_wmma<true, false, true, float><<<gqkv, 128, 0, stream>>>(
        ffn1, w216, out, b2, x2, M, DM, DFF);
}
